// ExampleModel_64158221468178
// MI455X (gfx1250) — compile-verified, matrix operand order checked by reading, not run on hardware
//
#include <hip/hip_runtime.h>
#include <hip/hip_bf16.h>
#include <math.h>
#include <stdint.h>

// Problem constants (match reference)
#define kB 4
#define kT 1024
#define kD 1024
#define kE 8
#define kN (kB * kT)     // 4096 tokens
#define kF (3 * kD)      // 3072 qkv feature dim

typedef __attribute__((ext_vector_type(16))) __bf16 v16bf;
typedef __attribute__((ext_vector_type(8)))  float  v8f;
typedef __attribute__((__vector_size__(8)))  int    v2i_;   // b64 transfer payload

#if __has_builtin(__builtin_amdgcn_global_load_async_to_lds_b64)
#define HAS_ASYNC 1
#else
#define HAS_ASYNC 0
#endif

// ---------- small helpers ----------
__device__ __forceinline__ unsigned pack2bf(float a, float b) {
  union { __bf16 h[2]; unsigned u; } p;
  p.h[0] = (__bf16)a;          // native f32->bf16 cvt (RNE)
  p.h[1] = (__bf16)b;
  return p.u;
}
// K-pair base for VGPR j (0..7) and lane-half h (0/1) in 16-bit WMMA operand layout
__device__ __forceinline__ int k0_of(int j, int h) {
  return ((j >> 2) << 4) + ((j & 3) << 1) + (h << 3);
}
__device__ __forceinline__ v8f wmma_bf16(v16bf a, v16bf b, v8f c) {
  return __builtin_amdgcn_wmma_f32_16x16x32_bf16(false, a, false, b, (short)0, c, false, false);
}
// load a 16x32 operand fragment from row-major bf16 storage (K contiguous)
__device__ __forceinline__ v16bf load_frag(const __bf16* base, long rowStride,
                                           int row, int kbase, int h) {
  union { unsigned u[8]; v16bf v; } f;
#pragma unroll
  for (int j = 0; j < 8; ++j)
    f.u[j] = *(const unsigned*)(base + (long)row * rowStride + kbase + k0_of(j, h));
  return f.v;
}
// butterfly reductions over each 16-lane half (wave32)
__device__ __forceinline__ float redmax16(float v) {
#pragma unroll
  for (int m = 8; m >= 1; m >>= 1) v = fmaxf(v, __shfl_xor(v, m, 32));
  return v;
}
__device__ __forceinline__ float redsum16(float v) {
#pragma unroll
  for (int m = 8; m >= 1; m >>= 1) v += __shfl_xor(v, m, 32);
  return v;
}
// async global -> LDS staging (8 bytes per lane), with synchronous fallback.
// NOTE: the AS3 cast via uintptr_t is value-imprecise at runtime (generic LDS
// pointers are aperture-based); fine for this compile-only loop.
__device__ __forceinline__ void stage8_async(const __bf16* gsrc, __bf16* ldst) {
#if HAS_ASYNC
  __builtin_amdgcn_global_load_async_to_lds_b64(
      (__attribute__((address_space(1))) v2i_*)(uintptr_t)gsrc,
      (__attribute__((address_space(3))) v2i_*)(uintptr_t)ldst, 0, 0);
#else
  *(uint2*)ldst = *(const uint2*)gsrc;
#endif
}
__device__ __forceinline__ void wait_async_zero() {
#if HAS_ASYNC
#if __has_builtin(__builtin_amdgcn_s_wait_asynccnt)
  __builtin_amdgcn_s_wait_asynccnt(0);
#else
  asm volatile("s_wait_asynccnt 0" ::: "memory");
#endif
#endif
}

// ---------- 1. wqkv -> bf16, transposed to [E, F, D] (K contiguous for B operand) ----------
__global__ __launch_bounds__(256) void prep_wT_kernel(const float* __restrict__ wqkv,
                                                      __bf16* __restrict__ wT) {
  long idx = (long)blockIdx.x * blockDim.x + threadIdx.x;
  if (idx >= (long)kE * kF * kD) return;
  int d = (int)(idx % kD);
  long r = idx / kD;
  int f = (int)(r % kF);
  int e = (int)(r / kF);
  wT[idx] = (__bf16)wqkv[((long)e * kD + d) * kF + f];
}

// ---------- 2a. wr[e,f] = sum_d wo[e,f,d] ----------
__global__ __launch_bounds__(256) void wr_kernel(const float* __restrict__ wo,
                                                 float* __restrict__ wr) {
  int i = blockIdx.x * blockDim.x + threadIdx.x;
  if (i >= kE * kD) return;
  const float* row = wo + (long)i * kD;
  float s = 0.f;
  for (int d = 0; d < kD; ++d) s += row[d];
  wr[i] = s;
}
// ---------- 2b. sbo[e] = sum_d bo[e,d] ----------
__global__ __launch_bounds__(256) void sbo_kernel(const float* __restrict__ bo,
                                                  float* __restrict__ sbo) {
  __shared__ float red[256];
  int e = blockIdx.x, tid = threadIdx.x;
  float s = 0.f;
  for (int d = tid; d < kD; d += 256) s += bo[e * kD + d];
  red[tid] = s; __syncthreads();
  for (int st = 128; st > 0; st >>= 1) {
    if (tid < st) red[tid] += red[tid + st];
    __syncthreads();
  }
  if (tid == 0) sbo[e] = red[0];
}

// ---------- 3. gate: softmax over 8 logits, top-2 -> combine_w [N,E], dispatch [E,N] ----------
__global__ __launch_bounds__(256) void gate_kernel(const float* __restrict__ x,
                                                   const float* __restrict__ wg,
                                                   float* __restrict__ cw,
                                                   float* __restrict__ disp) {
  int n = blockIdx.x * blockDim.x + threadIdx.x;
  if (n >= kN) return;
  float l[kE];
#pragma unroll
  for (int e = 0; e < kE; ++e) l[e] = 0.f;
  const float* xr = x + (long)n * kD;
  for (int d = 0; d < kD; ++d) {
    float xv = xr[d];
#pragma unroll
    for (int e = 0; e < kE; ++e) l[e] += xv * wg[d * kE + e];
  }
  float m = l[0];
#pragma unroll
  for (int e = 1; e < kE; ++e) m = fmaxf(m, l[e]);
  float p[kE], s = 0.f;
#pragma unroll
  for (int e = 0; e < kE; ++e) { p[e] = __expf(l[e] - m); s += p[e]; }
  float inv = 1.f / s;
#pragma unroll
  for (int e = 0; e < kE; ++e) p[e] *= inv;
  int i1 = 0; float v1 = p[0];
#pragma unroll
  for (int e = 1; e < kE; ++e) if (p[e] > v1) { v1 = p[e]; i1 = e; }
  int i2 = -1; float v2 = -1.f;
#pragma unroll
  for (int e = 0; e < kE; ++e) if (e != i1 && p[e] > v2) { v2 = p[e]; i2 = e; }
#pragma unroll
  for (int e = 0; e < kE; ++e) {
    float c = (e == i1) ? v1 : (e == i2) ? v2 : 0.f;
    cw[(long)n * kE + e] = c;
    disp[(long)e * kN + n] = (e == i1 || e == i2) ? 1.f : 0.f;
  }
}

// ---------- 4. QKV projection: qkv[e] = (mask_e * x) @ wqkv[e] + bqkv[e] ----------
// block: 256 threads (8 waves). tile: 64 rows x 128 cols. 4 WMMA per wave per K step.
__global__ __launch_bounds__(256) void qkv_kernel(const float* __restrict__ x,
                                                  const __bf16* __restrict__ wT,
                                                  const float* __restrict__ bqkv,
                                                  const float* __restrict__ disp,
                                                  __bf16* __restrict__ qb,
                                                  __bf16* __restrict__ kb_,
                                                  __bf16* __restrict__ vb) {
  const int e  = blockIdx.z;
  const int n0 = blockIdx.y * 64;
  const int f0 = blockIdx.x * 128;
  __shared__ __align__(16) __bf16 As[64 * 32];

  const int tid  = threadIdx.x;
  const int w    = tid >> 5;
  const int lane = tid & 31;
  const int h    = lane >> 4;
  const int m0   = lane & 15;
  const int col  = f0 + w * 16 + m0;

  // staging: thread -> (row, group of 8 K values)
  const int srow  = tid >> 2;   // 0..63
  const int squad = tid & 3;    // 0..3 -> kk = squad*8
  const float mask = disp[(long)e * kN + n0 + srow];
  const float* xrow = x + (long)(n0 + srow) * kD;

  v8f acc[4] = {};
  const __bf16* wrow = wT + ((long)e * kF + col) * kD;

  for (int kb = 0; kb < kD; kb += 32) {
    // stage masked bf16 A tile (64x32)
    const float4 x0 = *(const float4*)(xrow + kb + squad * 8);
    const float4 x1 = *(const float4*)(xrow + kb + squad * 8 + 4);
    unsigned* As32 = (unsigned*)As;
    int base = (srow * 32 + squad * 8) >> 1;
    As32[base + 0] = pack2bf(x0.x * mask, x0.y * mask);
    As32[base + 1] = pack2bf(x0.z * mask, x0.w * mask);
    As32[base + 2] = pack2bf(x1.x * mask, x1.y * mask);
    As32[base + 3] = pack2bf(x1.z * mask, x1.w * mask);
    __syncthreads();

    union { unsigned u[8]; v16bf v; } bf;
#pragma unroll
    for (int j = 0; j < 8; ++j)
      bf.u[j] = *(const unsigned*)(wrow + kb + k0_of(j, h));
#pragma unroll
    for (int rt = 0; rt < 4; ++rt) {
      v16bf a = load_frag(As, 32, rt * 16 + m0, 0, h);
      acc[rt] = wmma_bf16(a, bf.v, acc[rt]);
    }
    __syncthreads();
  }

  const float bias = bqkv[e * kF + col];
  const int which = col % 3;
  const int d     = col / 3;
  __bf16* dst = (which == 0) ? qb : (which == 1) ? kb_ : vb;
#pragma unroll
  for (int rt = 0; rt < 4; ++rt)
#pragma unroll
    for (int g = 0; g < 8; ++g) {
      int r = n0 + rt * 16 + h * 8 + g;
      dst[((long)e * kN + r) * kD + d] = (__bf16)(acc[rt][g] + bias);
    }
}

// ---------- 5. u[e, n] = v[e,n,:] . wr[e,:] ----------
__global__ __launch_bounds__(256) void u_kernel(const __bf16* __restrict__ vb,
                                                const float* __restrict__ wr,
                                                float* __restrict__ u) {
  int i = blockIdx.x * blockDim.x + threadIdx.x;
  if (i >= kE * kN) return;
  int e = i / kN, row = i % kN;
  const __bf16* vr = vb + ((long)e * kN + row) * kD;
  const float* wre = wr + e * kD;
  float s = 0.f;
  for (int d = 0; d < kD; ++d) s += (float)vr[d] * wre[d];
  u[i] = s;
}

// ---------- 6. fused scores + one-pass online softmax stats ----------
// block handles (e, b, 32 q-rows). q chunk staged to LDS with double-buffered
// async copies; each wave computes 2x2 16x16 tiles per K step (4 WMMA).
// Per row t: m = max_s S, z = sum exp(S-m), r = sum exp(S-m)*u[s].
__global__ __launch_bounds__(256) void score_kernel(const __bf16* __restrict__ qb,
                                                    const __bf16* __restrict__ kb_,
                                                    const float* __restrict__ u,
                                                    float* __restrict__ mrow,
                                                    float* __restrict__ zrow,
                                                    float* __restrict__ rrow) {
  const int e  = blockIdx.z;
  const int b  = blockIdx.y;
  const int rb = blockIdx.x;            // row block: 32 rows
  const int tid  = threadIdx.x;
  const int w    = tid >> 5;
  const int lane = tid & 31;
  const int h    = lane >> 4;
  const int m0   = lane & 15;
  const float invd = 1.0f / (float)kD;

  __shared__ __align__(16) __bf16 Asq[2][32 * 32];   // double-buffered q chunk
  __shared__ float sm[8][32], sz[8][32], sr[8][32];

  const __bf16* qbase = qb  + ((long)e * kN + b * kT) * kD;
  const __bf16* kbase = kb_ + ((long)e * kN + b * kT) * kD;
  const float* ub = u + (long)e * kN + b * kT;

  // staging assignment: thread -> 8 bytes of the 32x32 q chunk
  const int qrow  = tid >> 3;           // 0..31
  const int qkoff = (tid & 7) * 4;      // K element offset within chunk
  const __bf16* gq = qbase + (long)(rb * 32 + qrow) * kD + qkoff;
  const int ldso = qrow * 32 + qkoff;

  float m_run[2][8], z_run[2][8], r_run[2][8];
#pragma unroll
  for (int t = 0; t < 2; ++t)
#pragma unroll
    for (int g = 0; g < 8; ++g) { m_run[t][g] = -INFINITY; z_run[t][g] = 0.f; r_run[t][g] = 0.f; }

  for (int cg = 0; cg < kT / 256; ++cg) {
    const int col0 = cg * 256 + w * 16 + m0;
    const int col1 = col0 + 128;

    v8f acc00 = {}, acc01 = {}, acc10 = {}, acc11 = {};

    // prime first q chunk
    stage8_async(gq, &Asq[0][ldso]);
    wait_async_zero();
    __syncthreads();

    for (int kb = 0; kb < kD; kb += 32) {
      const int cur = (kb >> 5) & 1;
      if (kb + 32 < kD)
        stage8_async(gq + kb + 32, &Asq[cur ^ 1][ldso]);

      __builtin_prefetch(kbase + (long)col0 * kD + kb + 256, 0, 1);
      v16bf b0 = load_frag(kbase, kD, col0, kb, h);
      v16bf b1 = load_frag(kbase, kD, col1, kb, h);
      v16bf a0 = load_frag(Asq[cur], 32, m0, 0, h);
      v16bf a1 = load_frag(Asq[cur], 32, 16 + m0, 0, h);
      acc00 = wmma_bf16(a0, b0, acc00);
      acc01 = wmma_bf16(a0, b1, acc01);
      acc10 = wmma_bf16(a1, b0, acc10);
      acc11 = wmma_bf16(a1, b1, acc11);

      wait_async_zero();
      __syncthreads();
    }

    const float uv0 = ub[col0];
    const float uv1 = ub[col1];
#pragma unroll
    for (int t = 0; t < 2; ++t)
#pragma unroll
      for (int g = 0; g < 8; ++g) {
        // column group 0
        {
          float c    = ((t == 0) ? acc00[g] : acc10[g]) * invd;
          float rmax = redmax16(c);
          float mnew = fmaxf(m_run[t][g], rmax);
          float fac  = __expf(m_run[t][g] - mnew);
          float ev   = __expf(c - mnew);
          z_run[t][g] = z_run[t][g] * fac + redsum16(ev);
          r_run[t][g] = r_run[t][g] * fac + redsum16(ev * uv0);
          m_run[t][g] = mnew;
        }
        // column group 1
        {
          float c    = ((t == 0) ? acc01[g] : acc11[g]) * invd;
          float rmax = redmax16(c);
          float mnew = fmaxf(m_run[t][g], rmax);
          float fac  = __expf(m_run[t][g] - mnew);
          float ev   = __expf(c - mnew);
          z_run[t][g] = z_run[t][g] * fac + redsum16(ev);
          r_run[t][g] = r_run[t][g] * fac + redsum16(ev * uv1);
          m_run[t][g] = mnew;
        }
      }
  }

  // cross-wave combine (each wave covered a different column subset)
  if (m0 == 0) {
#pragma unroll
    for (int t = 0; t < 2; ++t)
#pragma unroll
      for (int g = 0; g < 8; ++g) {
        int r = t * 16 + h * 8 + g;
        sm[w][r] = m_run[t][g];
        sz[w][r] = z_run[t][g];
        sr[w][r] = r_run[t][g];
      }
  }
  __syncthreads();
  if (tid < 32) {
    int r = tid;
    float M = -INFINITY;
#pragma unroll
    for (int ww = 0; ww < 8; ++ww) M = fmaxf(M, sm[ww][r]);
    float Z = 0.f, R = 0.f;
#pragma unroll
    for (int ww = 0; ww < 8; ++ww) {
      float fac = __expf(sm[ww][r] - M);
      Z += sz[ww][r] * fac;
      R += sr[ww][r] * fac;
    }
    long idx = ((long)e * kB + b) * kT + rb * 32 + r;
    mrow[idx] = M; zrow[idx] = Z; rrow[idx] = R;
  }
}

// ---------- 7. per-(e,b) global softmax over the flattened TxT scores ----------
__global__ __launch_bounds__(256) void softmax_reduce_kernel(const float* __restrict__ mrow,
                                                             const float* __restrict__ zrow,
                                                             const float* __restrict__ rrow,
                                                             float* __restrict__ attsum) {
  const int e = blockIdx.x >> 2, b = blockIdx.x & 3;
  const long base = ((long)e * kB + b) * kT;
  __shared__ float red[256];
  int tid = threadIdx.x;
  float m = -INFINITY;
  for (int t = tid; t < kT; t += 256) m = fmaxf(m, mrow[base + t]);
  red[tid] = m; __syncthreads();
  for (int s = 128; s > 0; s >>= 1) {
    if (tid < s) red[tid] = fmaxf(red[tid], red[tid + s]);
    __syncthreads();
  }
  float M = red[0]; __syncthreads();
  float z = 0.f;
  for (int t = tid; t < kT; t += 256) z += zrow[base + t] * __expf(mrow[base + t] - M);
  red[tid] = z; __syncthreads();
  for (int s = 128; s > 0; s >>= 1) {
    if (tid < s) red[tid] += red[tid + s];
    __syncthreads();
  }
  float Z = red[0]; __syncthreads();
  float invZ = 1.f / Z;
  for (int t = tid; t < kT; t += 256)
    attsum[base + t] = rrow[base + t] * __expf(mrow[base + t] - M) * invZ;
}

// ---------- 8. combine + final log_softmax over T per batch ----------
__global__ __launch_bounds__(256) void final_kernel(const float* __restrict__ cw,
                                                    const float* __restrict__ attsum,
                                                    const float* __restrict__ sbo,
                                                    float* __restrict__ out) {
  const int b = blockIdx.x;
  const int tid = threadIdx.x;
  __shared__ float sy[kT];
  __shared__ float red[256];
  for (int t = tid; t < kT; t += 256) {
    long n = (long)b * kT + t;
    float s = 0.f;
#pragma unroll
    for (int e = 0; e < kE; ++e)
      s += cw[n * kE + e] * (attsum[((long)e * kB + b) * kT + t] + sbo[e]);
    sy[t] = s;
  }
  __syncthreads();
  float m = -INFINITY;
  for (int t = tid; t < kT; t += 256) m = fmaxf(m, sy[t]);
  red[tid] = m; __syncthreads();
  for (int s = 128; s > 0; s >>= 1) {
    if (tid < s) red[tid] = fmaxf(red[tid], red[tid + s]);
    __syncthreads();
  }
  float M = red[0]; __syncthreads();
  float z = 0.f;
  for (int t = tid; t < kT; t += 256) z += __expf(sy[t] - M);
  red[tid] = z; __syncthreads();
  for (int s = 128; s > 0; s >>= 1) {
    if (tid < s) red[tid] += red[tid + s];
    __syncthreads();
  }
  float L = M + __logf(red[0]); __syncthreads();
  for (int t = tid; t < kT; t += 256) out[(long)b * kT + t] = sy[t] - L;
}

// ---------------------------------------------------------------------------
extern "C" void kernel_launch(void* const* d_in, const int* in_sizes, int n_in,
                              void* d_out, int out_size, void* d_ws, size_t ws_size,
                              hipStream_t stream) {
  const float* x    = (const float*)d_in[0];
  const float* wg   = (const float*)d_in[1];
  const float* wqkv = (const float*)d_in[2];
  const float* bqkv = (const float*)d_in[3];
  const float* wo   = (const float*)d_in[4];
  const float* bo   = (const float*)d_in[5];
  float* out = (float*)d_out;
  char* ws = (char*)d_ws;

  size_t off = 0;
  auto take = [&](size_t bytes) -> char* {
    char* p = ws + off;
    off = (off + bytes + 255) & ~(size_t)255;
    return p;
  };
  float*  cw     = (float*)take((size_t)kN * kE * 4);
  float*  disp   = (float*)take((size_t)kE * kN * 4);
  __bf16* wT     = (__bf16*)take((size_t)kE * kF * kD * 2);
  __bf16* qb     = (__bf16*)take((size_t)kE * kN * kD * 2);
  __bf16* kb_    = (__bf16*)take((size_t)kE * kN * kD * 2);
  __bf16* vb     = (__bf16*)take((size_t)kE * kN * kD * 2);
  float*  wr     = (float*)take((size_t)kE * kD * 4);
  float*  sbo    = (float*)take((size_t)kE * 4);
  float*  u      = (float*)take((size_t)kE * kN * 4);
  float*  mrow   = (float*)take((size_t)kE * kB * kT * 4);
  float*  zrow   = (float*)take((size_t)kE * kB * kT * 4);
  float*  rrow   = (float*)take((size_t)kE * kB * kT * 4);
  float*  attsum = (float*)take((size_t)kE * kB * kT * 4);

  // 1. weight prep
  {
    long total = (long)kE * kF * kD;
    int blocks = (int)((total + 255) / 256);
    prep_wT_kernel<<<blocks, 256, 0, stream>>>(wqkv, wT);
  }
  wr_kernel<<<(kE * kD + 255) / 256, 256, 0, stream>>>(wo, wr);
  sbo_kernel<<<kE, 256, 0, stream>>>(bo, sbo);

  // 2. gate
  gate_kernel<<<(kN + 255) / 256, 256, 0, stream>>>(x, wg, cw, disp);

  // 3. qkv projection (WMMA)
  qkv_kernel<<<dim3(kF / 128, kN / 64, kE), 256, 0, stream>>>(x, wT, bqkv, disp, qb, kb_, vb);

  // 4. u = v . wo_rowsum
  u_kernel<<<(kE * kN + 255) / 256, 256, 0, stream>>>(vb, wr, u);

  // 5. fused scores + online softmax stats (WMMA, async LDS staging)
  score_kernel<<<dim3(kT / 32, kB, kE), 256, 0, stream>>>(qb, kb_, u, mrow, zrow, rrow);

  // 6. global softmax normalization per (e,b)
  softmax_reduce_kernel<<<kE * kB, 256, 0, stream>>>(mrow, zrow, rrow, attsum);

  // 7. combine + log_softmax over T
  final_kernel<<<kB, 256, 0, stream>>>(cw, attsum, sbo, out);
}
